// TransformerNet_18279380812408
// MI455X (gfx1250) — matrix operand order repaired and checked by reading, not compiled, not run on hardware
//
#include <hip/hip_runtime.h>
#include <hip/hip_bf16.h>

// ---------------------------------------------------------------------------
// GATv2 GNN for gfx1250 (MI455X, wave32).
// Dense GEMMs use v_wmma_f32_16x16x32_bf16 (bf16 inputs, fp32 accumulate).
// Edge softmax/scatter passes are fp32 with L2 atomics.
// N_NODES is a multiple of 16, so all WMMA M-tiles are full: no tail guards.
// ---------------------------------------------------------------------------

#define N_NODES   100000
#define HDIM      128
#define FIN       32
#define FOUT      16
#define NGRAPHS   64
#define NEG_SLOPE 0.2f

typedef __attribute__((ext_vector_type(16))) __bf16 v16bf;
typedef __attribute__((ext_vector_type(8)))  float  v8f;

// ----- monotonic float<->uint encoding for order-invariant atomic max -------
__device__ __forceinline__ unsigned fenc(float f) {
  unsigned u = __float_as_uint(f);
  return (u & 0x80000000u) ? ~u : (u | 0x80000000u);
}
__device__ __forceinline__ float fdec(unsigned e) {
  return __uint_as_float((e & 0x80000000u) ? (e & 0x7FFFFFFFu) : ~e);
}

// ---------------------------------------------------------------------------
// Pack fp32 weight W[K x 128] (row major) into bf16 WMMA B-fragment layout:
// out[((ks*8+n)*32 + lane)*16 + i] = bf16( W[k][n*16 + (lane&15)] )
//   lane<16 : k = ks*32 + (i<8 ? i : 16+i-8)      (K 0..7, 16..23)
//   lane>=16: k = ks*32 + 8 + (i<8 ? i : 16+i-8)  (K 8..15, 24..31)
// so each lane's v16bf load is 32B contiguous and fully coalesced.
// ---------------------------------------------------------------------------
__global__ void pack_weight_kernel(const float* __restrict__ W,
                                   __bf16* __restrict__ out, int ksteps) {
  int t = blockIdx.x * blockDim.x + threadIdx.x;
  int total = ksteps * 8 * 32 * 16;
  if (t >= total) return;
  int i    = t & 15;
  int lane = (t >> 4) & 31;
  int n    = (t >> 9) & 7;
  int ks   = t >> 12;
  int col  = n * 16 + (lane & 15);
  int k    = ks * 32 + ((lane >> 4) ? 8 : 0) + (i & 7) + ((i >> 3) << 4);
  out[t] = (__bf16)W[(size_t)k * HDIM + col];
}

// ---------------------------------------------------------------------------
// C[nrows x 128] = A[nrows x (KSTEPS*32)] @ B (pre-packed bf16) [+ bias]
// One wave per 16-row M tile; wave computes all 8 N tiles (16x128 out).
// nrows MUST be a multiple of 16 (true for this model: 100000 = 6250*16).
// HAS_BIAS is a compile-time flag: no runtime null-check branches.
// ---------------------------------------------------------------------------
template <int KSTEPS, bool HAS_BIAS>
__global__ void gemm_wmma_kernel(const float* __restrict__ A,
                                 const __bf16* __restrict__ Bp,
                                 const float* __restrict__ bias,
                                 float* __restrict__ C, int nrows) {
  const int lda  = KSTEPS * 32;
  int wave  = threadIdx.x >> 5;
  int lane  = threadIdx.x & 31;
  int mtile = blockIdx.x * 8 + wave;
  if (mtile >= (nrows >> 4)) return;     // wave-uniform exit
  int m0  = mtile << 4;
  int lo  = lane & 15;
  int hi  = lane >> 4;
  int row = m0 + lo;                     // always in range (nrows % 16 == 0)

  // A fragments: ISA 16-bit A 16x32 layout.
  v16bf afr[KSTEPS];
#pragma unroll
  for (int ks = 0; ks < KSTEPS; ++ks) {
    const float* ap = A + (size_t)row * lda + ks * 32 + hi * 8;
    float4 f0 = *(const float4*)(ap);
    float4 f1 = *(const float4*)(ap + 4);
    float4 f2 = *(const float4*)(ap + 16);
    float4 f3 = *(const float4*)(ap + 20);
    v16bf a;
    a[0]=(__bf16)f0.x;  a[1]=(__bf16)f0.y;  a[2]=(__bf16)f0.z;  a[3]=(__bf16)f0.w;
    a[4]=(__bf16)f1.x;  a[5]=(__bf16)f1.y;  a[6]=(__bf16)f1.z;  a[7]=(__bf16)f1.w;
    a[8]=(__bf16)f2.x;  a[9]=(__bf16)f2.y;  a[10]=(__bf16)f2.z; a[11]=(__bf16)f2.w;
    a[12]=(__bf16)f3.x; a[13]=(__bf16)f3.y; a[14]=(__bf16)f3.z; a[15]=(__bf16)f3.w;
    afr[ks] = a;
  }

  // Row base for this lane's half of the C tile (VGPR g -> row m0+g+hi*8).
  float* crow0 = C + (size_t)(m0 + hi * 8) * HDIM + lo;

#pragma unroll
  for (int n = 0; n < 8; ++n) {
    float bv = HAS_BIAS ? bias[n * 16 + lo] : 0.f;   // compile-time select
    v8f acc = {bv, bv, bv, bv, bv, bv, bv, bv};
#pragma unroll
    for (int ks = 0; ks < KSTEPS; ++ks) {
      v16bf bfg = *(const v16bf*)(Bp + ((size_t)(ks * 8 + n) * 512 + lane * 16));
      acc = __builtin_amdgcn_wmma_f32_16x16x32_bf16(
          /*neg_a=*/false, afr[ks], /*neg_b=*/false, bfg,
          /*c_mod=*/(short)0, acc, /*reuse_a=*/false, /*reuse_b=*/false);
    }
    // Unconditional stores: one base address + immediate offsets (g*512B).
    float* cr = crow0 + n * 16;
#pragma unroll
    for (int g = 0; g < 8; ++g) cr[g * HDIM] = acc[g];
  }
}

// out[n][j] = bias[j];  m_enc = enc(-inf);  s = 0
__global__ void init_layer_kernel(float* __restrict__ out,
                                  const float* __restrict__ bias,
                                  unsigned* __restrict__ m_enc,
                                  float* __restrict__ s, int nrows) {
  int t = blockIdx.x * blockDim.x + threadIdx.x;
  if (t < nrows * HDIM) out[t] = bias[t & (HDIM - 1)];
  if (t < nrows) { m_enc[t] = 0x007FFFFFu; s[t] = 0.f; }  // enc(-inf)
}

// Pass A: one wave per edge. e = leaky_relu(xl[src]+xr[dst]) . att ; seg-max.
__global__ void edge_score_kernel(const int* __restrict__ ei, int E, int Etot,
                                  const float* __restrict__ xl,
                                  const float* __restrict__ xr,
                                  const float* __restrict__ att,
                                  float* __restrict__ ebuf,
                                  unsigned* __restrict__ m_enc) {
  int gid  = blockIdx.x * blockDim.x + threadIdx.x;
  int edge = gid >> 5;
  int lane = gid & 31;
  if (edge >= Etot) return;
  int src, dst;
  if (edge < E) { src = ei[edge]; dst = ei[E + edge]; }
  else          { src = dst = edge - E; }              // self loops
  int j = lane * 4;
  float4 a = *(const float4*)(xl + (size_t)src * HDIM + j);
  float4 b = *(const float4*)(xr + (size_t)dst * HDIM + j);
  float4 w = *(const float4*)(att + j);
  float acc = 0.f, t;
  t = a.x + b.x; t = t > 0.f ? t : NEG_SLOPE * t; acc += t * w.x;
  t = a.y + b.y; t = t > 0.f ? t : NEG_SLOPE * t; acc += t * w.y;
  t = a.z + b.z; t = t > 0.f ? t : NEG_SLOPE * t; acc += t * w.z;
  t = a.w + b.w; t = t > 0.f ? t : NEG_SLOPE * t; acc += t * w.w;
#pragma unroll
  for (int off = 16; off > 0; off >>= 1) acc += __shfl_xor(acc, off, 32);
  if (lane == 0) {
    ebuf[edge] = acc;
    atomicMax(m_enc + dst, fenc(acc));
  }
}

// Pass B: p = exp(e - m[dst]); seg-sum into s.
__global__ void edge_exp_kernel(const int* __restrict__ ei, int E, int Etot,
                                float* __restrict__ ebuf,
                                const unsigned* __restrict__ m_enc,
                                float* __restrict__ s) {
  int edge = blockIdx.x * blockDim.x + threadIdx.x;
  if (edge >= Etot) return;
  int dst = (edge < E) ? ei[E + edge] : edge - E;
  float p = __expf(ebuf[edge] - fdec(m_enc[dst]));
  ebuf[edge] = p;
  atomicAdd(s + dst, p);
}

// Pass C: out[dst] += (p/s[dst]) * xl[src]; one wave per edge, 4 floats/lane.
__global__ void edge_scatter_kernel(const int* __restrict__ ei, int E, int Etot,
                                    const float* __restrict__ ebuf,
                                    const float* __restrict__ s,
                                    const float* __restrict__ xl,
                                    float* __restrict__ out) {
  int gid  = blockIdx.x * blockDim.x + threadIdx.x;
  int edge = gid >> 5;
  int lane = gid & 31;
  if (edge >= Etot) return;
  int src, dst;
  if (edge < E) { src = ei[edge]; dst = ei[E + edge]; }
  else          { src = dst = edge - E; }
  float alpha = ebuf[edge] / s[dst];
  int j = lane * 4;
  float4 v = *(const float4*)(xl + (size_t)src * HDIM + j);
  float* o = out + (size_t)dst * HDIM + j;
  atomicAdd(o + 0, alpha * v.x);
  atomicAdd(o + 1, alpha * v.y);
  atomicAdd(o + 2, alpha * v.z);
  atomicAdd(o + 3, alpha * v.w);
}

__global__ void zero_kernel(float* p, int n) {
  int t = blockIdx.x * blockDim.x + threadIdx.x;
  if (t < n) p[t] = 0.f;
}

// pooled[batch[i]][j] += h[ground[i]][j]
__global__ void pool_kernel(const float* __restrict__ h,
                            const int* __restrict__ ground,
                            const int* __restrict__ batch, int ng,
                            float* __restrict__ pooled) {
  int t = blockIdx.x * blockDim.x + threadIdx.x;
  int i = t >> 7, j = t & (HDIM - 1);
  if (i >= ng) return;
  atomicAdd(pooled + (size_t)batch[i] * HDIM + j, h[(size_t)ground[i] * HDIM + j]);
}

// out[g][j] = pooled[g] . out_W[:,j] + out_b[j]
__global__ void final_kernel(const float* __restrict__ pooled,
                             const float* __restrict__ outW,
                             const float* __restrict__ outB,
                             float* __restrict__ out) {
  int t = blockIdx.x * blockDim.x + threadIdx.x;
  if (t >= NGRAPHS * FOUT) return;
  int g = t >> 4, j = t & 15;
  float acc = outB[j];
  const float* pr = pooled + (size_t)g * HDIM;
#pragma unroll 8
  for (int k = 0; k < HDIM; ++k) acc += pr[k] * outW[k * FOUT + j];
  out[t] = acc;
}

// ---------------------------------------------------------------------------
extern "C" void kernel_launch(void* const* d_in, const int* in_sizes, int n_in,
                              void* d_out, int out_size, void* d_ws, size_t ws_size,
                              hipStream_t stream) {
  const float* x      = (const float*)d_in[0];
  const int*   ei_e   = (const int*)d_in[1];   // edge_index        [2,1M]
  const int*   ei_sg  = (const int*)d_in[2];   // subgraph_edge     [2,1M]
  const int*   ei_ns  = (const int*)d_in[3];   // node_subnode      [2,200k]
  const int*   ei_sn  = (const int*)d_in[4];   // subnode_node      [2,200k]
  const int*   ground = (const int*)d_in[5];
  const int*   batch  = (const int*)d_in[7];   // d_in[6] unused by reference
  const float* embW   = (const float*)d_in[8];
  const float* embB   = (const float*)d_in[9];
  const float* Wl     = (const float*)d_in[10];
  const float* Wr     = (const float*)d_in[11];
  const float* att    = (const float*)d_in[12];
  const float* bias   = (const float*)d_in[13];
  const float* outW   = (const float*)d_in[14];
  const float* outB   = (const float*)d_in[15];

  // ---- workspace carve-up (all offsets 256B aligned) ----
  const size_t S = (size_t)N_NODES * HDIM * sizeof(float);  // 51.2 MB
  char* w = (char*)d_ws;
  float*    hA     = (float*)(w);
  float*    hB     = (float*)(w + S);
  float*    xl     = (float*)(w + 2 * S);
  float*    xr     = (float*)(w + 3 * S);
  size_t off = 4 * S;
  float*    ebuf   = (float*)(w + off);    off += 4400128;   // <=1.1M edges
  unsigned* m_enc  = (unsigned*)(w + off); off += 400128;
  float*    sbuf   = (float*)(w + off);    off += 400128;
  float*    pooled = (float*)(w + off);    off += 32768;
  __bf16*   packEmb = (__bf16*)(w + off);  off += 8192;      // 1 kstep
  __bf16*   packW[8];
  for (int i = 0; i < 8; ++i) { packW[i] = (__bf16*)(w + off); off += 32768; }

  // ---- pack weights to bf16 WMMA fragment layout (cheap, every call) ----
  pack_weight_kernel<<<16, 256, 0, stream>>>(embW, packEmb, 1);
  for (int l = 0; l < 4; ++l) {
    pack_weight_kernel<<<64, 256, 0, stream>>>(Wl + (size_t)l * HDIM * HDIM, packW[2 * l], 4);
    pack_weight_kernel<<<64, 256, 0, stream>>>(Wr + (size_t)l * HDIM * HDIM, packW[2 * l + 1], 4);
  }

  const int mtiles = N_NODES / 16;                 // 6250 (exact)
  const int gblocks = (mtiles + 7) / 8;            // 8 waves per block

  // ---- h = x @ emb_W + emb_b  (K=32 -> 1 WMMA k-step) ----
  gemm_wmma_kernel<1, true><<<gblocks, 256, 0, stream>>>(x, packEmb, embB, hA, N_NODES);

  const int* eis[4] = {ei_e, ei_ns, ei_sg, ei_sn};
  int Es[4] = {in_sizes[1] / 2, in_sizes[3] / 2, in_sizes[2] / 2, in_sizes[4] / 2};

  float* hcur = hA;
  float* hnxt = hB;
  for (int l = 0; l < 4; ++l) {
    gemm_wmma_kernel<4, false><<<gblocks, 256, 0, stream>>>(hcur, packW[2 * l],
                                                            nullptr, xl, N_NODES);
    gemm_wmma_kernel<4, false><<<gblocks, 256, 0, stream>>>(hcur, packW[2 * l + 1],
                                                            nullptr, xr, N_NODES);

    int tot = N_NODES * HDIM;
    init_layer_kernel<<<(tot + 255) / 256, 256, 0, stream>>>(hnxt, bias + l * HDIM,
                                                             m_enc, sbuf, N_NODES);
    int Etot = Es[l] + N_NODES;  // + self loops
    int ewave_blocks = (int)(((long long)Etot * 32 + 255) / 256);
    edge_score_kernel<<<ewave_blocks, 256, 0, stream>>>(eis[l], Es[l], Etot, xl, xr,
                                                        att + l * HDIM, ebuf, m_enc);
    edge_exp_kernel<<<(Etot + 255) / 256, 256, 0, stream>>>(eis[l], Es[l], Etot,
                                                            ebuf, m_enc, sbuf);
    edge_scatter_kernel<<<ewave_blocks, 256, 0, stream>>>(eis[l], Es[l], Etot,
                                                          ebuf, sbuf, xl, hnxt);
    float* t = hcur; hcur = hnxt; hnxt = t;
  }

  // ---- global_add_pool over ground nodes, then final linear ----
  zero_kernel<<<(NGRAPHS * HDIM + 255) / 256, 256, 0, stream>>>(pooled, NGRAPHS * HDIM);
  int NG = in_sizes[5];
  pool_kernel<<<(int)(((long long)NG * HDIM + 255) / 256), 256, 0, stream>>>(
      hcur, ground, batch, NG, pooled);
  final_kernel<<<(NGRAPHS * FOUT + 255) / 256, 256, 0, stream>>>(pooled, outW, outB,
                                                                 (float*)d_out);
}
